// GCN_EL_CP_38714835206182
// MI455X (gfx1250) — matrix operand order repaired and checked by reading, not compiled, not run on hardware
//
#include <hip/hip_runtime.h>
#include <math.h>

#define NNODES 50000
#define NEDGES 800000
#define NPOOL  400000
#define NCOMM  1000

typedef __attribute__((ext_vector_type(16))) __bf16 v16bf;
typedef __attribute__((ext_vector_type(8)))  __bf16 v8bf;
typedef __attribute__((ext_vector_type(8)))  float  v8f;

// ---------------------------------------------------------------------------
// WMMA helpers (gfx1250, wave32). A: 16x32 bf16, B: 32x16 bf16, C/D: 16x16 f32
// A layout per lane (half = lane>>4, m = lane&15): K = 8*half+i (i<8),
// K = 16+8*half+(i-8) (i>=8)  -> two contiguous 16B loads from row-major A.
// ---------------------------------------------------------------------------
static __device__ __forceinline__ v16bf make_a(const __bf16* __restrict__ row,
                                               int colbase, int half) {
  v8bf lo = *(const v8bf*)(row + colbase + 8 * half);
  v8bf hi = *(const v8bf*)(row + colbase + 16 + 8 * half);
  v16bf r;
#pragma unroll
  for (int i = 0; i < 8; ++i) { r[i] = lo[i]; r[8 + i] = hi[i]; }
  return r;
}

static __device__ __forceinline__ v8f wmma_bf16(v16bf a, v16bf b, v8f c) {
  return __builtin_amdgcn_wmma_f32_16x16x32_bf16(false, a, false, b, (short)0,
                                                 c, false, false);
}

// ---------------------------------------------------------------------------
// Weight repack: W[K,N] f32 row-major -> bf16 WMMA B-tiles.
// Tile t=(kt,nt): element index ((t*32)+lane)*16+i holds
// W[kt*32 + (i<8 ? 8*(lane>>4)+i : 16+8*(lane>>4)+(i-8))][nt*16 + (lane&15)]  (one 32B load per lane).
// ---------------------------------------------------------------------------
__global__ void k_prep(const float* __restrict__ W, int ktiles, int ntiles,
                       __bf16* __restrict__ Bp) {
  int idx = blockIdx.x * blockDim.x + threadIdx.x;
  int total = ktiles * ntiles * 512;
  if (idx >= total) return;
  int t = idx >> 9, r = idx & 511;
  int lane = r >> 4, i = r & 15;
  int kt = t / ntiles, nt = t - kt * ntiles;
  int half = lane >> 4, n = lane & 15;
  int K = kt * 32 + (i < 8 ? 8 * half + i : 16 + 8 * half + (i - 8));
  int N = ntiles * 16;
  Bp[idx] = (__bf16)W[(size_t)K * N + nt * 16 + n];
}

// Wl1 [512,256] folded: p=(r1,r1) => use Wl1[0:256]+Wl1[256:512], K=256,N=256
__global__ void k_prep_lin1(const float* __restrict__ W, __bf16* __restrict__ Bp) {
  int idx = blockIdx.x * blockDim.x + threadIdx.x;
  const int ntiles = 16, N = 256;
  int total = 8 * 16 * 512;
  if (idx >= total) return;
  int t = idx >> 9, r = idx & 511;
  int lane = r >> 4, i = r & 15;
  int kt = t / ntiles, nt = t - kt * ntiles;
  int half = lane >> 4, n = lane & 15;
  int K = kt * 32 + (i < 8 ? 8 * half + i : 16 + 8 * half + (i - 8));
  int col = nt * 16 + n;
  Bp[idx] = (__bf16)(W[(size_t)K * N + col] + W[(size_t)(K + 256) * N + col]);
}

// ---------------------------------------------------------------------------
// Tiny-K scalar kernels
// ---------------------------------------------------------------------------
__global__ void k_embed(const float* __restrict__ x, const float* __restrict__ W1,
                        const float* __restrict__ b1, const float* __restrict__ W2,
                        const float* __restrict__ b2, __bf16* __restrict__ hcat) {
  int idx = blockIdx.x * blockDim.x + threadIdx.x;
  if (idx >= NNODES * 128) return;
  int nrow = idx >> 7, j = idx & 127;
  const float* xr = x + (size_t)nrow * 20;
  float s;
  if (j < 64) {
    s = b1[j];
#pragma unroll
    for (int k = 0; k < 8; ++k) s += xr[k] * W1[k * 64 + j];
  } else {
    int jj = j - 64;
    s = b2[jj];
#pragma unroll
    for (int k = 0; k < 12; ++k) s += xr[8 + k] * W2[k * 64 + jj];
  }
  hcat[idx] = (__bf16)fmaxf(s, 0.f);
}

__global__ void k_ea(const float* __restrict__ ea, const float* __restrict__ We,
                     const float* __restrict__ be, __bf16* __restrict__ out) {
  long long idx = (long long)blockIdx.x * blockDim.x + threadIdx.x;
  if (idx >= (long long)NEDGES * 64) return;
  int e = (int)(idx >> 6), j = (int)(idx & 63);
  const float* r = ea + (size_t)e * 8;
  float s = be[j];
#pragma unroll
  for (int k = 0; k < 8; ++k) s += r[k] * We[k * 64 + j];
  out[idx] = (__bf16)fmaxf(s, 0.f);
}

__global__ void k_count(const int* __restrict__ idx, int n, float* __restrict__ c) {
  int i = blockIdx.x * blockDim.x + threadIdx.x;
  if (i < n) atomicAdd(&c[idx[i]], 1.0f);
}

__global__ void k_divmean(const float* __restrict__ s, const float* __restrict__ c,
                          __bf16* __restrict__ o, int rows, int ld) {
  long long idx = (long long)blockIdx.x * blockDim.x + threadIdx.x;
  if (idx >= (long long)rows * ld) return;
  int r = (int)(idx / ld);
  o[idx] = (__bf16)(s[idx] / fmaxf(c[r], 1.0f));
}

__global__ void k_final(const float* __restrict__ g, const float* __restrict__ Wl2,
                        const float* __restrict__ bl2, float* __restrict__ out) {
  int c = blockIdx.x * blockDim.x + threadIdx.x;
  if (c >= NCOMM) return;
  const float* r = g + (size_t)c * 256;
  float s = bl2[0];
  for (int k = 0; k < 256; ++k) s += r[k] * Wl2[k];
  out[c] = s;
}

// ---------------------------------------------------------------------------
// Generic WMMA GEMM: C = act(A @ W + b). Wave handles 16 x (NBLK*16) tile,
// A tile loaded once per K-step and reused across NBLK B tiles.
// ---------------------------------------------------------------------------
template <int ACT, int NBLK, bool SF, bool SB>
__global__ __launch_bounds__(256) void k_gemm(
    const __bf16* __restrict__ A, int lda, const __bf16* __restrict__ Bp,
    const float* __restrict__ bias, int mtiles, int ktiles, int ntiles,
    float* __restrict__ Cf, __bf16* __restrict__ Cb, int ldc) {
  int wave = (int)((blockIdx.x * blockDim.x + threadIdx.x) >> 5);
  int lane = threadIdx.x & 31;
  int nblocks = ntiles / NBLK;
  if (wave >= mtiles * nblocks) return;      // wave-uniform
  int mt = wave % mtiles;
  int nt0 = (wave / mtiles) * NBLK;
  int half = lane >> 4, mn = lane & 15;
  const __bf16* arow = A + (size_t)(mt * 16 + mn) * lda;
  v8f acc[NBLK] = {};
  for (int kt = 0; kt < ktiles; ++kt) {
    v16bf a = make_a(arow, kt * 32, half);
#pragma unroll
    for (int j = 0; j < NBLK; ++j) {
      v16bf b = *(const v16bf*)(Bp + (((size_t)kt * ntiles + nt0 + j) * 32 + lane) * 16);
      acc[j] = wmma_bf16(a, b, acc[j]);
    }
  }
#pragma unroll
  for (int j = 0; j < NBLK; ++j) {
    int chan = (nt0 + j) * 16 + mn;
    float bv = bias ? bias[chan] : 0.f;
#pragma unroll
    for (int v = 0; v < 8; ++v) {
      int row = mt * 16 + v + 8 * half;
      float val = acc[j][v] + bv;
      if (ACT == 1) val = fmaxf(val, 0.f);
      size_t off = (size_t)row * ldc + chan;
      if (SF) Cf[off] = val;
      if (SB) Cb[off] = (__bf16)val;
    }
  }
}

// ---------------------------------------------------------------------------
// mask = relu([h[src] | h[dst] | ea] @ Wel + bel)   K=320, N=128
// A rows gathered per K-tile; no [E,320] buffer is materialized.
// ---------------------------------------------------------------------------
template <int NBLK>
__global__ __launch_bounds__(256) void k_mask(
    const __bf16* __restrict__ hb, const __bf16* __restrict__ eab,
    const int* __restrict__ src, const int* __restrict__ dst,
    const __bf16* __restrict__ Bp, const float* __restrict__ bias,
    __bf16* __restrict__ maskb) {
  const int etiles = NEDGES / 16, ntiles = 8;
  int wave = (int)((blockIdx.x * blockDim.x + threadIdx.x) >> 5);
  int lane = threadIdx.x & 31;
  if (wave >= etiles * (ntiles / NBLK)) return;
  int et = wave % etiles;
  int nt0 = (wave / etiles) * NBLK;
  int half = lane >> 4, mn = lane & 15;
  int e = et * 16 + mn;
  const __bf16* rs = hb + (size_t)src[e] * 128;
  const __bf16* rd = hb + (size_t)dst[e] * 128;
  const __bf16* re = eab + (size_t)e * 64;
  v8f acc[NBLK] = {};
  for (int kt = 0; kt < 10; ++kt) {
    const __bf16* row;
    int cb;
    if (kt < 4)      { row = rs; cb = kt * 32; }
    else if (kt < 8) { row = rd; cb = (kt - 4) * 32; }
    else             { row = re; cb = (kt - 8) * 32; }
    v16bf a = make_a(row, cb, half);
#pragma unroll
    for (int j = 0; j < NBLK; ++j) {
      v16bf b = *(const v16bf*)(Bp + (((size_t)kt * ntiles + nt0 + j) * 32 + lane) * 16);
      acc[j] = wmma_bf16(a, b, acc[j]);
    }
  }
#pragma unroll
  for (int j = 0; j < NBLK; ++j) {
    int chan = (nt0 + j) * 16 + mn;
    float bv = bias[chan];
#pragma unroll
    for (int v = 0; v < 8; ++v) {
      int row = et * 16 + v + 8 * half;
      maskb[(size_t)row * 128 + chan] = (__bf16)fmaxf(acc[j][v] + bv, 0.f);
    }
  }
}

// ---------------------------------------------------------------------------
// Fused elconv: gate = sigmoid(mask@Wm+bm); scatter h[src]*gate into agg[dst]
// ---------------------------------------------------------------------------
template <int NBLK>
__global__ __launch_bounds__(256) void k_elconv(
    const __bf16* __restrict__ maskb, const __bf16* __restrict__ Bp,
    const float* __restrict__ bm, const float* __restrict__ hf,
    const int* __restrict__ src, const int* __restrict__ dst,
    float* __restrict__ agg) {
  const int etiles = NEDGES / 16, ntiles = 8;
  int wave = (int)((blockIdx.x * blockDim.x + threadIdx.x) >> 5);
  int lane = threadIdx.x & 31;
  if (wave >= etiles * (ntiles / NBLK)) return;
  int et = wave % etiles;
  int nt0 = (wave / etiles) * NBLK;
  int half = lane >> 4, mn = lane & 15;
  const __bf16* arow = maskb + (size_t)(et * 16 + mn) * 128;
  v8f acc[NBLK] = {};
  for (int kt = 0; kt < 4; ++kt) {
    v16bf a = make_a(arow, kt * 32, half);
#pragma unroll
    for (int j = 0; j < NBLK; ++j) {
      v16bf b = *(const v16bf*)(Bp + (((size_t)kt * ntiles + nt0 + j) * 32 + lane) * 16);
      acc[j] = wmma_bf16(a, b, acc[j]);
    }
  }
#pragma unroll
  for (int j = 0; j < NBLK; ++j) {
    int chan = (nt0 + j) * 16 + mn;
    float bv = bm[chan];
#pragma unroll
    for (int v = 0; v < 8; ++v) {
      int e = et * 16 + v + 8 * half;
      float g = 1.0f / (1.0f + __expf(-(acc[j][v] + bv)));
      float msg = hf[(size_t)src[e] * 128 + chan] * g;   // gather: L2 resident
      atomicAdd(agg + (size_t)dst[e] * 128 + chan, msg); // segment sum
    }
  }
}

// ---------------------------------------------------------------------------
// Fused pool: xp = relu([h[pi]|h[pj]] @ Wam + bam); segment-sum by community
// ---------------------------------------------------------------------------
template <int NBLK>
__global__ __launch_bounds__(256) void k_pool(
    const __bf16* __restrict__ hb, const int* __restrict__ pi,
    const int* __restrict__ pj, const int* __restrict__ comm,
    const __bf16* __restrict__ Bp, const float* __restrict__ bias,
    float* __restrict__ r1sum) {
  const int ptiles = NPOOL / 16, ntiles = 16;
  int wave = (int)((blockIdx.x * blockDim.x + threadIdx.x) >> 5);
  int lane = threadIdx.x & 31;
  if (wave >= ptiles * (ntiles / NBLK)) return;
  int pt = wave % ptiles;
  int nt0 = (wave / ptiles) * NBLK;
  int half = lane >> 4, mn = lane & 15;
  int p = pt * 16 + mn;
  const __bf16* ri = hb + (size_t)pi[p] * 128;
  const __bf16* rj = hb + (size_t)pj[p] * 128;
  v8f acc[NBLK] = {};
  for (int kt = 0; kt < 8; ++kt) {
    const __bf16* row = (kt < 4) ? ri : rj;
    v16bf a = make_a(row, (kt & 3) * 32, half);
#pragma unroll
    for (int j = 0; j < NBLK; ++j) {
      v16bf b = *(const v16bf*)(Bp + (((size_t)kt * ntiles + nt0 + j) * 32 + lane) * 16);
      acc[j] = wmma_bf16(a, b, acc[j]);
    }
  }
#pragma unroll
  for (int j = 0; j < NBLK; ++j) {
    int chan = (nt0 + j) * 16 + mn;
    float bv = bias[chan];
#pragma unroll
    for (int v = 0; v < 8; ++v) {
      int pr = pt * 16 + v + 8 * half;
      float val = fmaxf(acc[j][v] + bv, 0.f);
      atomicAdd(r1sum + (size_t)comm[pr] * 256 + chan, val);
    }
  }
}

// ---------------------------------------------------------------------------
static inline unsigned cdiv(size_t a, size_t b) { return (unsigned)((a + b - 1) / b); }

extern "C" void kernel_launch(void* const* d_in, const int* in_sizes, int n_in,
                              void* d_out, int out_size, void* d_ws, size_t ws_size,
                              hipStream_t stream) {
  (void)in_sizes; (void)n_in; (void)out_size; (void)ws_size;
  const float* x    = (const float*)d_in[0];
  const float* eatt = (const float*)d_in[1];
  const int*   eidx = (const int*)d_in[2];
  const int*   comm = (const int*)d_in[3];
  const int*   pedg = (const int*)d_in[4];
  const float* We  = (const float*)d_in[5];  const float* be  = (const float*)d_in[6];
  const float* W1  = (const float*)d_in[7];  const float* b1  = (const float*)d_in[8];
  const float* W2  = (const float*)d_in[9];  const float* b2  = (const float*)d_in[10];
  const float* W3  = (const float*)d_in[11]; const float* b3  = (const float*)d_in[12];
  const float* Wel = (const float*)d_in[13]; const float* bel = (const float*)d_in[14];
  const float* Wm1 = (const float*)d_in[15]; const float* bm1 = (const float*)d_in[16];
  const float* Wc1 = (const float*)d_in[17]; const float* bc1 = (const float*)d_in[18];
  const float* Wm2 = (const float*)d_in[19]; const float* bm2 = (const float*)d_in[20];
  const float* Wc2 = (const float*)d_in[21]; const float* bc2 = (const float*)d_in[22];
  const float* Wam = (const float*)d_in[23]; const float* bam = (const float*)d_in[24];
  const float* Wl1 = (const float*)d_in[25]; const float* bl1 = (const float*)d_in[26];
  const float* Wl2 = (const float*)d_in[27]; const float* bl2 = (const float*)d_in[28];

  const int* src = eidx;  const int* dstI = eidx + NEDGES;
  const int* pi  = pedg;  const int* pj   = pedg + NPOOL;

  // ---- workspace carve-up (deterministic, ~400 MB) ----
  char* base = (char*)d_ws; size_t off = 0;
  auto carve = [&](size_t bytes) -> char* {
    char* p = base + off; off += (bytes + 255) & ~(size_t)255; return p;
  };
  __bf16* W3p  = (__bf16*)carve((size_t)4 * 8 * 512 * 2);
  __bf16* Welp = (__bf16*)carve((size_t)10 * 8 * 512 * 2);
  __bf16* Wm1p = (__bf16*)carve((size_t)4 * 8 * 512 * 2);
  __bf16* Wc1p = (__bf16*)carve((size_t)4 * 8 * 512 * 2);
  __bf16* Wm2p = (__bf16*)carve((size_t)4 * 8 * 512 * 2);
  __bf16* Wc2p = (__bf16*)carve((size_t)4 * 8 * 512 * 2);
  __bf16* Wamp = (__bf16*)carve((size_t)8 * 16 * 512 * 2);
  __bf16* Wl1p = (__bf16*)carve((size_t)8 * 16 * 512 * 2);
  __bf16* hcat = (__bf16*)carve((size_t)NNODES * 128 * 2);
  __bf16* hb   = (__bf16*)carve((size_t)NNODES * 128 * 2);
  float*  hf   = (float*) carve((size_t)NNODES * 128 * 4);
  __bf16* eab  = (__bf16*)carve((size_t)NEDGES * 64 * 2);
  __bf16* maskb= (__bf16*)carve((size_t)NEDGES * 128 * 2);
  float*  agg  = (float*) carve((size_t)NNODES * 128 * 4);
  __bf16* aggb = (__bf16*)carve((size_t)NNODES * 128 * 2);
  float*  cntd = (float*) carve((size_t)NNODES * 4);
  float*  r1s  = (float*) carve((size_t)NCOMM * 256 * 4);
  float*  cntc = (float*) carve((size_t)NCOMM * 4);
  __bf16* rcb  = (__bf16*)carve((size_t)1008 * 256 * 2);   // C padded to 16
  float*  gf   = (float*) carve((size_t)1008 * 256 * 4);

  // ---- weight repack to WMMA B layout ----
  auto prep = [&](const float* W, int kt, int nt, __bf16* dp) {
    int total = kt * nt * 512;
    k_prep<<<cdiv(total, 256), 256, 0, stream>>>(W, kt, nt, dp);
  };
  prep(W3, 4, 8, W3p);   prep(Wel, 10, 8, Welp);
  prep(Wm1, 4, 8, Wm1p); prep(Wc1, 4, 8, Wc1p);
  prep(Wm2, 4, 8, Wm2p); prep(Wc2, 4, 8, Wc2p);
  prep(Wam, 8, 16, Wamp);
  k_prep_lin1<<<cdiv(8 * 16 * 512, 256), 256, 0, stream>>>(Wl1, Wl1p);

  // ---- zero accumulators (capturable memset nodes) ----
  hipMemsetAsync(cntd, 0, (size_t)NNODES * 4, stream);
  hipMemsetAsync(cntc, 0, (size_t)NCOMM * 4, stream);
  hipMemsetAsync(agg,  0, (size_t)NNODES * 128 * 4, stream);
  hipMemsetAsync(r1s,  0, (size_t)NCOMM * 256 * 4, stream);
  hipMemsetAsync(rcb,  0, (size_t)1008 * 256 * 2, stream);

  // ---- degree counts (shared by both convs / both pools) ----
  k_count<<<cdiv(NEDGES, 256), 256, 0, stream>>>(dstI, NEDGES, cntd);
  k_count<<<cdiv(NPOOL, 256), 256, 0, stream>>>(comm, NPOOL, cntc);

  // ---- node + edge embeddings (tiny K: scalar) ----
  k_embed<<<cdiv((size_t)NNODES * 128, 256), 256, 0, stream>>>(x, W1, b1, W2, b2, hcat);
  k_ea<<<cdiv((size_t)NEDGES * 64, 256), 256, 0, stream>>>(eatt, We, be, eab);

  // ---- h = relu(hcat @ W3 + b3)  [50000,128] ----
  {
    unsigned waves = 3125 * (8 / 4);
    k_gemm<1, 4, true, true><<<cdiv((size_t)waves * 32, 256), 256, 0, stream>>>(
        hcat, 128, W3p, b3, 3125, 4, 8, hf, hb, 128);
  }

  // ---- mask = relu([h[src]|h[dst]|ea] @ Wel + bel)  [800000,128] ----
  {
    unsigned waves = (NEDGES / 16) * (8 / 4);
    k_mask<4><<<cdiv((size_t)waves * 32, 256), 256, 0, stream>>>(
        hb, eab, src, dstI, Welp, bel, maskb);
  }

  // ---- elconv1: gate GEMM + gather + scatter-sum ----
  {
    unsigned waves = (NEDGES / 16) * (8 / 4);
    k_elconv<4><<<cdiv((size_t)waves * 32, 256), 256, 0, stream>>>(
        maskb, Wm1p, bm1, hf, src, dstI, agg);
  }
  k_divmean<<<cdiv((size_t)NNODES * 128, 256), 256, 0, stream>>>(agg, cntd, aggb, NNODES, 128);
  {  // h = relu(aggmean @ Wc1 + bc1)
    unsigned waves = 3125 * 2;
    k_gemm<1, 4, true, true><<<cdiv((size_t)waves * 32, 256), 256, 0, stream>>>(
        aggb, 128, Wc1p, bc1, 3125, 4, 8, hf, hb, 128);
  }

  // ---- pool1: attn MLP + community segment-sum ----
  {
    unsigned waves = (NPOOL / 16) * (16 / 4);
    k_pool<4><<<cdiv((size_t)waves * 32, 256), 256, 0, stream>>>(
        hb, pi, pj, comm, Wamp, bam, r1s);
  }

  // ---- elconv2 (uses hf from conv1, same mask) ----
  hipMemsetAsync(agg, 0, (size_t)NNODES * 128 * 4, stream);
  {
    unsigned waves = (NEDGES / 16) * 2;
    k_elconv<4><<<cdiv((size_t)waves * 32, 256), 256, 0, stream>>>(
        maskb, Wm2p, bm2, hf, src, dstI, agg);
  }
  k_divmean<<<cdiv((size_t)NNODES * 128, 256), 256, 0, stream>>>(agg, cntd, aggb, NNODES, 128);
  {  // h = relu(aggmean @ Wc2 + bc2)  (only bf16 needed downstream)
    unsigned waves = 3125 * 2;
    k_gemm<1, 4, false, true><<<cdiv((size_t)waves * 32, 256), 256, 0, stream>>>(
        aggb, 128, Wc2p, bc2, 3125, 4, 8, nullptr, hb, 128);
  }

  // ---- pool2 accumulates into same r1s (identical community counts) ----
  {
    unsigned waves = (NPOOL / 16) * 4;
    k_pool<4><<<cdiv((size_t)waves * 32, 256), 256, 0, stream>>>(
        hb, pi, pj, comm, Wamp, bam, r1s);
  }

  // ---- rc = (sum1+sum2)/count ; g = relu(rc @ (Wl1a+Wl1b) + bl1) ----
  k_divmean<<<cdiv((size_t)NCOMM * 256, 256), 256, 0, stream>>>(r1s, cntc, rcb, NCOMM, 256);
  {
    unsigned waves = 63 * (16 / 4);   // 1008 rows padded (pad rows zeroed)
    k_gemm<1, 4, true, false><<<cdiv((size_t)waves * 32, 256), 256, 0, stream>>>(
        rcb, 256, Wl1p, bl1, 63, 8, 16, gf, nullptr, 256);
  }

  // ---- out = g @ Wl2 + bl2 ----
  k_final<<<cdiv(NCOMM, 256), 256, 0, stream>>>(gf, Wl2, bl2, (float*)d_out);
}
